// ModernTransformerBlock_5377299054802
// MI455X (gfx1250) — compile-verified, matrix operand order checked by reading, not run on hardware
//
#include <hip/hip_runtime.h>
#include <hip/hip_bf16.h>
#include <math.h>

// ---------------------------------------------------------------------------
// ModernTransformerBlock for MI455X (gfx1250, wave32, WMMA)
// B=4, L=4096, E=1024, H=16, D=64, DFF=4096
// Big GEMMs: bf16 A/B via v_wmma_f32_16x16x32_bf16, fp32 accumulate.
// SwiGLU gate/up GEMMs fused into one kernel (dual accumulators) to kill the
// 0.5 GB fp32 gate-buffer roundtrip. bf16 weights+activations stay L2-resident.
// Attention state (cosformer linear attention) in fp32 VALU, L split into
// 16 deterministic partial reductions for parallelism.
// ---------------------------------------------------------------------------

typedef __attribute__((ext_vector_type(8)))  __bf16 bf16x8;
typedef __attribute__((ext_vector_type(16))) __bf16 bf16x16;
typedef __attribute__((ext_vector_type(8)))  float  f32x8;

static constexpr int Bb   = 4;
static constexpr int Ll   = 4096;
static constexpr int Ee   = 1024;
static constexpr int Hh   = 16;
static constexpr int Dd   = 64;
static constexpr int DFF  = 4096;
static constexpr int Mrows = Bb * Ll;          // 16384
static constexpr int NCHUNK = 16;              // L-split for kv state
static constexpr int CLEN   = Ll / NCHUNK;     // 256 rows per chunk

__device__ __forceinline__ __hip_bfloat16 f2bf(float f) { return __float2bfloat16(f); }

// ---------------------------------------------------------------------------
// Weight transpose + fp32 -> bf16 cast:  W[K][N] (row major) -> Wt[N][K]
// ---------------------------------------------------------------------------
__global__ void __launch_bounds__(256)
transpose_cast_kernel(const float* __restrict__ W, __hip_bfloat16* __restrict__ Wt,
                      int K, int N)
{
    __shared__ float tile[32][33];
    int n0 = blockIdx.x * 32;
    int k0 = blockIdx.y * 32;
    for (int i = threadIdx.y; i < 32; i += 8)
        tile[i][threadIdx.x] = W[(size_t)(k0 + i) * N + n0 + threadIdx.x];
    __syncthreads();
    for (int i = threadIdx.y; i < 32; i += 8)
        Wt[(size_t)(n0 + i) * K + k0 + threadIdx.x] = f2bf(tile[threadIdx.x][i]);
}

// ---------------------------------------------------------------------------
// RMSNorm: out_bf16[row][e] = x[row][e] * w[e] * rsqrt(mean(x^2) + 1e-6)
// ---------------------------------------------------------------------------
__global__ void __launch_bounds__(256)
rmsnorm_kernel(const float* __restrict__ x, const float* __restrict__ w,
               __hip_bfloat16* __restrict__ out, int E)
{
    __shared__ float red[256];
    int row = blockIdx.x;
    const float* xr = x + (size_t)row * E;
    float p = 0.f;
    for (int e = threadIdx.x; e < E; e += 256) { float v = xr[e]; p = fmaf(v, v, p); }
    red[threadIdx.x] = p;
    __syncthreads();
    for (int s = 128; s > 0; s >>= 1) {
        if (threadIdx.x < s) red[threadIdx.x] += red[threadIdx.x + s];
        __syncthreads();
    }
    float rstd = rsqrtf(red[0] / (float)E + 1e-6f);
    for (int e = threadIdx.x; e < E; e += 256)
        out[(size_t)row * E + e] = f2bf(xr[e] * w[e] * rstd);
}

// ---------------------------------------------------------------------------
// WMMA fragment load for 16-bit 16x32 operand, K-contiguous in memory.
// lanes 0-15 -> row = lane,    K = {0..7, 16..23} + k0
// lanes16-31 -> row = lane-16, K = {8..15,24..31} + k0
// ---------------------------------------------------------------------------
__device__ __forceinline__ bf16x16
load_frag(const __hip_bfloat16* __restrict__ X, int ld, int base, int k0, int lane)
{
    const __bf16* p = reinterpret_cast<const __bf16*>(X)
                    + (size_t)(base + (lane & 15)) * ld + k0 + ((lane >> 4) << 3);
    bf16x8 lo = *reinterpret_cast<const bf16x8*>(p);
    bf16x8 hi = *reinterpret_cast<const bf16x8*>(p + 16);
    return __builtin_shufflevector(lo, hi, 0,1,2,3,4,5,6,7,8,9,10,11,12,13,14,15);
}

// ---------------------------------------------------------------------------
// GEMM: C[M][N] = A[M][K](bf16) * Bt[N][K](bf16)^T  with fused epilogues
//   mode 0: v = acc (+bias)
//   mode 1: v = acc (+bias) + res[idx]           (residual add)
// output: outB (bf16) if non-null else outF (fp32)
// 256 threads = 8 waves as 4(M) x 2(N); wave tile 32x64 (2x4 frags);
// block tile 128(M) x 128(N). grid = (N/128, M/128).
// ---------------------------------------------------------------------------
__global__ void __launch_bounds__(256)
gemm_bf16_wmma(const __hip_bfloat16* __restrict__ A,
               const __hip_bfloat16* __restrict__ Bt,
               const float* __restrict__ bias,
               const float* __restrict__ res,
               float* __restrict__ outF,
               __hip_bfloat16* __restrict__ outB,
               int M, int N, int K, int mode)
{
    int tid  = threadIdx.x;
    int lane = tid & 31;
    int w    = tid >> 5;       // 0..7
    int wm   = w >> 1;         // 0..3
    int wn   = w & 1;          // 0..1
    int row0 = blockIdx.y * 128 + wm * 32;
    int col0 = blockIdx.x * 128 + wn * 64;

    f32x8 acc[2][4] = {};

    for (int k0 = 0; k0 < K; k0 += 32) {
        bf16x16 a0 = load_frag(A,  K, row0,      k0, lane);
        bf16x16 a1 = load_frag(A,  K, row0 + 16, k0, lane);
        bf16x16 b0 = load_frag(Bt, K, col0,      k0, lane);
        bf16x16 b1 = load_frag(Bt, K, col0 + 16, k0, lane);
        bf16x16 b2 = load_frag(Bt, K, col0 + 32, k0, lane);
        bf16x16 b3 = load_frag(Bt, K, col0 + 48, k0, lane);
        acc[0][0] = __builtin_amdgcn_wmma_f32_16x16x32_bf16(false, a0, false, b0,
                        (short)0, acc[0][0], false, false);
        acc[0][1] = __builtin_amdgcn_wmma_f32_16x16x32_bf16(false, a0, false, b1,
                        (short)0, acc[0][1], false, false);
        acc[0][2] = __builtin_amdgcn_wmma_f32_16x16x32_bf16(false, a0, false, b2,
                        (short)0, acc[0][2], false, false);
        acc[0][3] = __builtin_amdgcn_wmma_f32_16x16x32_bf16(false, a0, false, b3,
                        (short)0, acc[0][3], false, false);
        acc[1][0] = __builtin_amdgcn_wmma_f32_16x16x32_bf16(false, a1, false, b0,
                        (short)0, acc[1][0], false, false);
        acc[1][1] = __builtin_amdgcn_wmma_f32_16x16x32_bf16(false, a1, false, b1,
                        (short)0, acc[1][1], false, false);
        acc[1][2] = __builtin_amdgcn_wmma_f32_16x16x32_bf16(false, a1, false, b2,
                        (short)0, acc[1][2], false, false);
        acc[1][3] = __builtin_amdgcn_wmma_f32_16x16x32_bf16(false, a1, false, b3,
                        (short)0, acc[1][3], false, false);
    }

    // Epilogue. C/D layout: VGPR t -> (m = t or t+8 by lane half, n = lane&15).
    #pragma unroll
    for (int i = 0; i < 2; ++i) {
        #pragma unroll
        for (int j = 0; j < 4; ++j) {
            int rbase = row0 + i * 16 + ((lane < 16) ? 0 : 8);
            int c     = col0 + j * 16 + (lane & 15);
            float bv  = bias ? bias[c] : 0.f;
            #pragma unroll
            for (int t = 0; t < 8; ++t) {
                int r = rbase + t;
                size_t idx = (size_t)r * N + c;
                float v = acc[i][j][t] + bv;
                if (mode == 1) v += res[idx];
                if (outB) outB[idx] = f2bf(v);
                else      outF[idx] = v;
            }
        }
    }
}

// ---------------------------------------------------------------------------
// Fused SwiGLU GEMM: act[M][N] = silu(A @ Bg^T) * (A @ Bu^T)   (bf16 out)
// Dual accumulator sets over shared A fragments; no fp32 gate buffer.
// 256 threads = 8 waves as 4(M) x 2(N); wave tile 32x32 per matrix;
// block tile 128(M) x 64(N). grid = (N/64, M/128).
// ---------------------------------------------------------------------------
__global__ void __launch_bounds__(256)
gemm_swiglu_fused(const __hip_bfloat16* __restrict__ A,
                  const __hip_bfloat16* __restrict__ Bg,
                  const __hip_bfloat16* __restrict__ Bu,
                  __hip_bfloat16* __restrict__ outB,
                  int M, int N, int K)
{
    int tid  = threadIdx.x;
    int lane = tid & 31;
    int w    = tid >> 5;       // 0..7
    int wm   = w >> 1;         // 0..3
    int wn   = w & 1;          // 0..1
    int row0 = blockIdx.y * 128 + wm * 32;
    int col0 = blockIdx.x * 64  + wn * 32;

    f32x8 accG[2][2] = {};
    f32x8 accU[2][2] = {};

    for (int k0 = 0; k0 < K; k0 += 32) {
        bf16x16 a0  = load_frag(A,  K, row0,      k0, lane);
        bf16x16 a1  = load_frag(A,  K, row0 + 16, k0, lane);
        bf16x16 bg0 = load_frag(Bg, K, col0,      k0, lane);
        bf16x16 bg1 = load_frag(Bg, K, col0 + 16, k0, lane);
        bf16x16 bu0 = load_frag(Bu, K, col0,      k0, lane);
        bf16x16 bu1 = load_frag(Bu, K, col0 + 16, k0, lane);
        accG[0][0] = __builtin_amdgcn_wmma_f32_16x16x32_bf16(false, a0, false, bg0,
                        (short)0, accG[0][0], false, false);
        accG[0][1] = __builtin_amdgcn_wmma_f32_16x16x32_bf16(false, a0, false, bg1,
                        (short)0, accG[0][1], false, false);
        accG[1][0] = __builtin_amdgcn_wmma_f32_16x16x32_bf16(false, a1, false, bg0,
                        (short)0, accG[1][0], false, false);
        accG[1][1] = __builtin_amdgcn_wmma_f32_16x16x32_bf16(false, a1, false, bg1,
                        (short)0, accG[1][1], false, false);
        accU[0][0] = __builtin_amdgcn_wmma_f32_16x16x32_bf16(false, a0, false, bu0,
                        (short)0, accU[0][0], false, false);
        accU[0][1] = __builtin_amdgcn_wmma_f32_16x16x32_bf16(false, a0, false, bu1,
                        (short)0, accU[0][1], false, false);
        accU[1][0] = __builtin_amdgcn_wmma_f32_16x16x32_bf16(false, a1, false, bu0,
                        (short)0, accU[1][0], false, false);
        accU[1][1] = __builtin_amdgcn_wmma_f32_16x16x32_bf16(false, a1, false, bu1,
                        (short)0, accU[1][1], false, false);
    }

    #pragma unroll
    for (int i = 0; i < 2; ++i) {
        #pragma unroll
        for (int j = 0; j < 2; ++j) {
            int rbase = row0 + i * 16 + ((lane < 16) ? 0 : 8);
            int c     = col0 + j * 16 + (lane & 15);
            #pragma unroll
            for (int t = 0; t < 8; ++t) {
                int r = rbase + t;
                size_t idx = (size_t)r * N + c;
                float g = accG[i][j][t];
                float u = accU[i][j][t];
                float v = (g / (1.f + expf(-g))) * u;    // silu(gate) * up
                outB[idx] = f2bf(v);
            }
        }
    }
}

// ---------------------------------------------------------------------------
// Cosformer state, partial over an L-chunk: for (b,h,chunk) compute
//   pKv_cos[d][e] = sum_{l in chunk} k_cos[l][d]*v[l][e]   (and sin)
//   pKsum[d]      = sum_{l in chunk} k_cos[l][d]           (and sin)
// grid (B*H, NCHUNK); 256 threads; LDS-staged rows.
// ---------------------------------------------------------------------------
#define KVCH 8
__global__ void __launch_bounds__(256)
kv_state_partial_kernel(const float* __restrict__ qkv,
                        const float* __restrict__ fcos, const float* __restrict__ fsin,
                        float* __restrict__ pKvC, float* __restrict__ pKvS,
                        float* __restrict__ pKsC, float* __restrict__ pKsS)
{
    __shared__ float kc[KVCH][64], ks[KVCH][64], vv[KVCH][64];
    int bh    = blockIdx.x;
    int chunk = blockIdx.y;
    int b  = bh >> 4;
    int hh = bh & 15;
    int tid  = threadIdx.x;
    int lrow = tid >> 5;           // 0..7  (row within staging chunk)
    int j    = tid & 31;           // rope pair index 0..31
    int d    = tid >> 2;           // 0..63
    int e0   = (tid & 3) << 4;     // 0,16,32,48

    float accC[16], accS[16];
    #pragma unroll
    for (int i = 0; i < 16; ++i) { accC[i] = 0.f; accS[i] = 0.f; }
    float sumC = 0.f, sumS = 0.f;

    int lbeg = chunk * CLEN;
    for (int l0 = lbeg; l0 < lbeg + CLEN; l0 += KVCH) {
        int l = l0 + lrow;
        const float* rowp = qkv + (size_t)(b * Ll + l) * (3 * Ee);
        float kr = rowp[Ee + hh * Dd + 2 * j];
        float ki = rowp[Ee + hh * Dd + 2 * j + 1];
        float c  = fcos[l * 32 + j];
        float s  = fsin[l * 32 + j];
        float xr = kr * c - ki * s;
        float xi = kr * s + ki * c;
        xr = fmaxf(xr, 0.f); xi = fmaxf(xi, 0.f);
        float ang = 1.57079632679489662f * (float)l * (1.0f / (float)Ll);
        float cw = cosf(ang), sw = sinf(ang);
        kc[lrow][2 * j]     = xr * cw;  ks[lrow][2 * j]     = xr * sw;
        kc[lrow][2 * j + 1] = xi * cw;  ks[lrow][2 * j + 1] = xi * sw;
        vv[lrow][2 * j]     = rowp[2 * Ee + hh * Dd + 2 * j];
        vv[lrow][2 * j + 1] = rowp[2 * Ee + hh * Dd + 2 * j + 1];
        __syncthreads();

        #pragma unroll
        for (int r = 0; r < KVCH; ++r) {
            float kcv = kc[r][d], ksv = ks[r][d];
            #pragma unroll
            for (int i = 0; i < 16; ++i) {
                float vx = vv[r][e0 + i];
                accC[i] = fmaf(kcv, vx, accC[i]);
                accS[i] = fmaf(ksv, vx, accS[i]);
            }
        }
        if (tid < 64) {
            #pragma unroll
            for (int r = 0; r < KVCH; ++r) { sumC += kc[r][tid]; sumS += ks[r][tid]; }
        }
        __syncthreads();
    }

    size_t pbase = ((size_t)chunk * (Bb * Hh) + bh) * (Dd * Dd) + (size_t)d * Dd + e0;
    #pragma unroll
    for (int i = 0; i < 16; ++i) { pKvC[pbase + i] = accC[i]; pKvS[pbase + i] = accS[i]; }
    if (tid < 64) {
        size_t sbase = ((size_t)chunk * (Bb * Hh) + bh) * Dd + tid;
        pKsC[sbase] = sumC; pKsS[sbase] = sumS;
    }
}

// ---------------------------------------------------------------------------
// Deterministic reduction over the NCHUNK partials: out[i] = sum_c part[c*n+i]
// ---------------------------------------------------------------------------
__global__ void __launch_bounds__(256)
reduce_chunks_kernel(const float* __restrict__ part, float* __restrict__ out, int n)
{
    int i = blockIdx.x * 256 + threadIdx.x;
    if (i >= n) return;
    float s = 0.f;
    #pragma unroll
    for (int c = 0; c < NCHUNK; ++c) s += part[(size_t)c * n + i];
    out[i] = s;
}

// ---------------------------------------------------------------------------
// Context: out[l][e] = (q_cos[l]·kv_cos[:,e] + q_sin[l]·kv_sin[:,e])
//                      / (q_cos[l]·ksum_cos + q_sin[l]·ksum_sin + 1e-4)
// bf16 out at attn[(b*L+l)*E + h*64 + e]. grid (B*H, L/128), 256 threads.
// ---------------------------------------------------------------------------
__global__ void __launch_bounds__(256)
context_kernel(const float* __restrict__ qkv,
               const float* __restrict__ fcos, const float* __restrict__ fsin,
               const float* __restrict__ kvC, const float* __restrict__ kvS,
               const float* __restrict__ ksC, const float* __restrict__ ksS,
               __hip_bfloat16* __restrict__ attn)
{
    __shared__ float kvc[Dd * Dd], kvs[Dd * Dd];
    __shared__ float ksc[Dd], kss[Dd];
    __shared__ float qc[4][Dd], qs[4][Dd];

    int bh = blockIdx.x;
    int b  = bh >> 4;
    int hh = bh & 15;
    int tid = threadIdx.x;

    for (int i = tid; i < Dd * Dd; i += 256) {
        kvc[i] = kvC[(size_t)bh * (Dd * Dd) + i];
        kvs[i] = kvS[(size_t)bh * (Dd * Dd) + i];
    }
    if (tid < Dd) { ksc[tid] = ksC[bh * 64 + tid]; kss[tid] = ksS[bh * 64 + tid]; }
    __syncthreads();

    for (int it = 0; it < 32; ++it) {
        int l0 = blockIdx.y * 128 + it * 4;
        if (tid < 128) {
            int row = tid >> 5, j = tid & 31;
            int l = l0 + row;
            const float* rowp = qkv + (size_t)(b * Ll + l) * (3 * Ee) + hh * Dd;
            float qr = rowp[2 * j], qi = rowp[2 * j + 1];
            float c = fcos[l * 32 + j], s = fsin[l * 32 + j];
            float xr = qr * c - qi * s;
            float xi = qr * s + qi * c;
            xr = fmaxf(xr, 0.f); xi = fmaxf(xi, 0.f);
            float ang = 1.57079632679489662f * (float)l * (1.0f / (float)Ll);
            float cw = cosf(ang), sw = sinf(ang);
            qc[row][2 * j]     = xr * cw;  qs[row][2 * j]     = xr * sw;
            qc[row][2 * j + 1] = xi * cw;  qs[row][2 * j + 1] = xi * sw;
        }
        __syncthreads();
        {
            int row = tid >> 6, e = tid & 63;
            int l = l0 + row;
            float ctx = 0.f, nrm = 0.f;
            #pragma unroll 8
            for (int d = 0; d < Dd; ++d) {
                float a  = qc[row][d];
                float bb = qs[row][d];
                ctx = fmaf(a,  kvc[d * 64 + e], ctx);
                ctx = fmaf(bb, kvs[d * 64 + e], ctx);
                nrm = fmaf(a,  ksc[d], nrm);
                nrm = fmaf(bb, kss[d], nrm);
            }
            float o = ctx / (nrm + 1e-4f);
            attn[(size_t)(b * Ll + l) * Ee + hh * Dd + e] = f2bf(o);
        }
        __syncthreads();
    }
}

// ---------------------------------------------------------------------------
// Host launcher
// ---------------------------------------------------------------------------
extern "C" void kernel_launch(void* const* d_in, const int* in_sizes, int n_in,
                              void* d_out, int out_size, void* d_ws, size_t ws_size,
                              hipStream_t stream) {
    (void)in_sizes; (void)n_in; (void)out_size; (void)ws_size;

    const float* x    = (const float*)d_in[0];
    const float* wqkv = (const float*)d_in[1];
    const float* bqkv = (const float*)d_in[2];
    const float* wo   = (const float*)d_in[3];
    const float* bo   = (const float*)d_in[4];
    const float* n1w  = (const float*)d_in[5];
    const float* n2w  = (const float*)d_in[6];
    const float* wg   = (const float*)d_in[7];
    const float* wu   = (const float*)d_in[8];
    const float* wd   = (const float*)d_in[9];
    const float* fc   = (const float*)d_in[10];
    const float* fs   = (const float*)d_in[11];
    float* out = (float*)d_out;

    char* ws = (char*)d_ws;
    size_t off = 0;
    auto alloc = [&](size_t bytes) -> char* {
        char* p = ws + off;
        off += bytes;
        off = (off + 255) & ~(size_t)255;
        return p;
    };

    typedef __hip_bfloat16 hbf;
    hbf*   wqkvT = (hbf*)alloc((size_t)(3 * Ee) * Ee * 2);      // [3072][1024]
    hbf*   woT   = (hbf*)alloc((size_t)Ee * Ee * 2);            // [1024][1024]
    hbf*   wgT   = (hbf*)alloc((size_t)DFF * Ee * 2);           // [4096][1024]
    hbf*   wuT   = (hbf*)alloc((size_t)DFF * Ee * 2);           // [4096][1024]
    hbf*   wdT   = (hbf*)alloc((size_t)Ee * DFF * 2);           // [1024][4096]
    hbf*   hbfp  = (hbf*)alloc((size_t)Mrows * Ee * 2);         // rmsnorm1 out
    float* qkvf  = (float*)alloc((size_t)Mrows * 3 * Ee * 4);   // qkv fp32
    float* pKvC  = (float*)alloc((size_t)NCHUNK * Bb * Hh * Dd * Dd * 4);
    float* pKvS  = (float*)alloc((size_t)NCHUNK * Bb * Hh * Dd * Dd * 4);
    float* pKsC  = (float*)alloc((size_t)NCHUNK * Bb * Hh * Dd * 4);
    float* pKsS  = (float*)alloc((size_t)NCHUNK * Bb * Hh * Dd * 4);
    float* kvC   = (float*)alloc((size_t)Bb * Hh * Dd * Dd * 4);
    float* kvS   = (float*)alloc((size_t)Bb * Hh * Dd * Dd * 4);
    float* ksC   = (float*)alloc((size_t)Bb * Hh * Dd * 4);
    float* ksS   = (float*)alloc((size_t)Bb * Hh * Dd * 4);
    hbf*   attnb = (hbf*)alloc((size_t)Mrows * Ee * 2);         // attention out bf16
    float* x2f   = (float*)alloc((size_t)Mrows * Ee * 4);       // x + attn proj
    hbf*   h2bf  = (hbf*)alloc((size_t)Mrows * Ee * 2);         // rmsnorm2 out
    hbf*   actbf = (hbf*)alloc((size_t)Mrows * DFF * 2);        // silu(gate)*up

    dim3 tb(32, 8);
    // Weight transpose+cast to bf16 [N][K]
    transpose_cast_kernel<<<dim3((3 * Ee) / 32, Ee / 32), tb, 0, stream>>>(wqkv, wqkvT, Ee, 3 * Ee);
    transpose_cast_kernel<<<dim3(Ee / 32, Ee / 32),       tb, 0, stream>>>(wo,   woT,   Ee, Ee);
    transpose_cast_kernel<<<dim3(DFF / 32, Ee / 32),      tb, 0, stream>>>(wg,   wgT,   Ee, DFF);
    transpose_cast_kernel<<<dim3(DFF / 32, Ee / 32),      tb, 0, stream>>>(wu,   wuT,   Ee, DFF);
    transpose_cast_kernel<<<dim3(Ee / 32, DFF / 32),      tb, 0, stream>>>(wd,   wdT,   DFF, Ee);

    // h = rms_norm(x, norm1_w)  (bf16)
    rmsnorm_kernel<<<Mrows, 256, 0, stream>>>(x, n1w, hbfp, Ee);

    // qkv = h @ wqkv + bqkv  (fp32 out)
    gemm_bf16_wmma<<<dim3((3 * Ee) / 128, Mrows / 128), 256, 0, stream>>>(
        hbfp, wqkvT, bqkv, nullptr, qkvf, nullptr, Mrows, 3 * Ee, Ee, 0);

    // cosformer linear-attention state (chunked partials + deterministic reduce)
    kv_state_partial_kernel<<<dim3(Bb * Hh, NCHUNK), 256, 0, stream>>>(
        qkvf, fc, fs, pKvC, pKvS, pKsC, pKsS);
    {
        int nkv = Bb * Hh * Dd * Dd;   // 262144
        int nks = Bb * Hh * Dd;        // 4096
        reduce_chunks_kernel<<<(nkv + 255) / 256, 256, 0, stream>>>(pKvC, kvC, nkv);
        reduce_chunks_kernel<<<(nkv + 255) / 256, 256, 0, stream>>>(pKvS, kvS, nkv);
        reduce_chunks_kernel<<<(nks + 255) / 256, 256, 0, stream>>>(pKsC, ksC, nks);
        reduce_chunks_kernel<<<(nks + 255) / 256, 256, 0, stream>>>(pKsS, ksS, nks);
    }
    context_kernel<<<dim3(Bb * Hh, Ll / 128), 256, 0, stream>>>(
        qkvf, fc, fs, kvC, kvS, ksC, ksS, attnb);

    // x2 = x + attn @ wo + bo
    gemm_bf16_wmma<<<dim3(Ee / 128, Mrows / 128), 256, 0, stream>>>(
        attnb, woT, bo, x, x2f, nullptr, Mrows, Ee, Ee, 1);

    // h2 = rms_norm(x2, norm2_w)
    rmsnorm_kernel<<<Mrows, 256, 0, stream>>>(x2f, n2w, h2bf, Ee);

    // act = silu(h2 @ w_gate) * (h2 @ w_up)   (fused, bf16 out)
    gemm_swiglu_fused<<<dim3(DFF / 64, Mrows / 128), 256, 0, stream>>>(
        h2bf, wgT, wuT, actbf, Mrows, DFF, Ee);

    // out = x2 + act @ w_down
    gemm_bf16_wmma<<<dim3(Ee / 128, Mrows / 128), 256, 0, stream>>>(
        actbf, wdT, nullptr, x2f, out, nullptr, Mrows, Ee, DFF, 1);
}
